// SingleHeadAttention_38362647888505
// MI455X (gfx1250) — matrix operand order repaired and checked by reading, not compiled, not run on hardware
//
#include <hip/hip_runtime.h>
#include <stdint.h>

typedef unsigned short u16_t;
typedef __attribute__((ext_vector_type(16))) __bf16 v16bf;
typedef __attribute__((ext_vector_type(8)))  float  v8f;

#define MDIM 1024
#define HDIM 128
#define SLEN 4096
#define NROWS 16384  // 4 * 4096 total rows

// 16x32 bf16 A-fragment / 32x16 bf16 B-fragment: 8 VGPRs = 16 bf16 per lane.
union FragB {
  uint4 u[2];
  v16bf v;
};

__device__ __forceinline__ u16_t f2bf(float f) {
  uint32_t u = __float_as_uint(f);
  u += 0x7FFFu + ((u >> 16) & 1u);  // round-to-nearest-even
  return (u16_t)(u >> 16);
}

__device__ __forceinline__ uint32_t pk2bf(float lo, float hi) {
  return (uint32_t)f2bf(lo) | ((uint32_t)f2bf(hi) << 16);
}

__device__ __forceinline__ v8f bf_wmma(v16bf a, v16bf b, v8f c) {
  // D = A(16x32) * B(32x16) + C, f32 accumulate
  return __builtin_amdgcn_wmma_f32_16x16x32_bf16(false, a, false, b, (short)0, c,
                                                 false, false);
}

// Hard scheduling fence: nothing moves across. Used to split "issue all loads"
// from "consume with WMMA" so the scheduler can't sink loads next to their
// consumers (which collapses them onto one register octet and forces
// s_wait_loadcnt 0 before every WMMA).
__device__ __forceinline__ void sched_split() { __builtin_amdgcn_sched_barrier(0); }

// ---------------- Stage 0a: E f32 -> bf16 ----------------
__global__ __launch_bounds__(256) void cvt_e_kernel(const float* __restrict__ E,
                                                    u16_t* __restrict__ Ebf) {
  size_t i = ((size_t)blockIdx.x * 256 + threadIdx.x) * 4;
  float4 f = *(const float4*)(E + i);
  ushort4 o;
  o.x = f2bf(f.x); o.y = f2bf(f.y); o.z = f2bf(f.z); o.w = f2bf(f.w);
  *(ushort4*)(Ebf + i) = o;
}

// ---------------- Stage 0b: W f32 -> bf16 (Wq pre-scaled by 1/sqrt(H)) ------
__global__ __launch_bounds__(256) void cvt_w_kernel(const float* __restrict__ Wk,
                                                    const float* __restrict__ Wq,
                                                    const float* __restrict__ Wv,
                                                    u16_t* __restrict__ Wbf) {
  int i = blockIdx.x * 256 + threadIdx.x;  // 131072 elements per matrix
  const float scale = 0.08838834764831845f;  // 1/sqrt(128)
  Wbf[i] = f2bf(Wk[i]);
  Wbf[HDIM * MDIM + i] = f2bf(Wq[i] * scale);
  Wbf[2 * HDIM * MDIM + i] = f2bf(Wv[i]);
}

// ---------------- Stage 1: QKV projection ----------------
// Grid: 1024 blocks (16 rows each). Block: 96 threads = 3 waves.
// Wave w computes target w (0=K row-major, 1=Q row-major scaled, 2=V transposed).
__global__ __launch_bounds__(96) void proj_kernel(const u16_t* __restrict__ Ebf,
                                                  const u16_t* __restrict__ Wbf,
                                                  u16_t* __restrict__ Qbf,
                                                  u16_t* __restrict__ Kbf,
                                                  u16_t* __restrict__ Vt) {
  const int wv = threadIdx.x >> 5;   // 0,1,2 -> K,Q,V
  const int lane = threadIdx.x & 31;
  const int m = lane & 15;           // row (A) / col (B) index
  const int lh = lane >> 4;          // lane half
  const int rowBase = blockIdx.x * 16;
  const u16_t* W = Wbf + (size_t)wv * HDIM * MDIM;

  v8f acc[8] = {};
  const u16_t* arow = Ebf + (size_t)(rowBase + m) * MDIM + lh * 16;
  const u16_t* wrow = W + (size_t)m * MDIM + lh * 16;

#pragma unroll 2
  for (int k0 = 0; k0 < MDIM; k0 += 32) {
    // Phase 1: issue ALL loads of this k-step (one clause burst, 18 loads)
    FragB a;
    a.u[0] = *(const uint4*)(arow + k0);        // K = k0 + lh*16 + 0..7
    a.u[1] = *(const uint4*)(arow + k0 + 8);    // K = k0 + lh*16 + 8..15
    FragB bw[8];
#pragma unroll
    for (int nt = 0; nt < 8; ++nt) {
      const u16_t* brow = wrow + (size_t)nt * (16 * MDIM) + k0;
      bw[nt].u[0] = *(const uint4*)(brow);      // K = k0 + lh*16 + 0..7
      bw[nt].u[1] = *(const uint4*)(brow + 8);  // K = k0 + lh*16 + 8..15
    }
    sched_split();  // loads stay above; next iter's loads may overlap WMMAs below
    // Phase 2: 8 WMMAs (partial loadcnt waits; next iteration loads interleave)
#pragma unroll
    for (int nt = 0; nt < 8; ++nt)
      acc[nt] = bf_wmma(a.v, bw[nt].v, acc[nt]);
  }

  if (wv == 2) {  // V stored transposed: Vt[h * NROWS + row]; pack row pairs
#pragma unroll
    for (int nt = 0; nt < 8; ++nt)
#pragma unroll
      for (int v = 0; v < 8; v += 2)
        *(uint32_t*)(Vt + (size_t)(nt * 16 + m) * NROWS + rowBase + v + 8 * lh) =
            pk2bf(acc[nt][v], acc[nt][v + 1]);
  } else {
    u16_t* dst = (wv == 0) ? Kbf : Qbf;
#pragma unroll
    for (int nt = 0; nt < 8; ++nt)
#pragma unroll
      for (int v = 0; v < 8; ++v)
        dst[(size_t)(rowBase + v + 8 * lh) * HDIM + nt * 16 + m] = f2bf(acc[nt][v]);
  }
}

// ---------------- Stage 2: causal flash attention ----------------
// Grid: 4 batches * 64 q-blocks = 256 blocks. Block: 128 threads = 4 waves.
// Wave w owns 16 query rows; online softmax over key tiles of 32.
__global__ __launch_bounds__(128) void attn_kernel(const u16_t* __restrict__ Qbf,
                                                   const u16_t* __restrict__ Kbf,
                                                   const u16_t* __restrict__ Vt,
                                                   float* __restrict__ out) {
  __shared__ __align__(16) u16_t plds[4][16 * 40];  // per-wave P transpose buffer
  const int wv = threadIdx.x >> 5;
  const int lane = threadIdx.x & 31;
  const int n = lane & 15;
  const int lh = lane >> 4;
  const int b = blockIdx.x >> 6;
  const int qblk = blockIdx.x & 63;
  const int qLocal = qblk * 64 + wv * 16;
  const size_t rowG = (size_t)b * SLEN + qLocal;

  // Persistent Q A-fragments (16 rows x 128 head dims = 4 chunks of K=32)
  FragB qf[4];
#pragma unroll
  for (int c = 0; c < 4; ++c) {
    const u16_t* p = Qbf + (rowG + n) * HDIM + c * 32 + lh * 16;
    qf[c].u[0] = *(const uint4*)(p);
    qf[c].u[1] = *(const uint4*)(p + 8);
  }

  v8f o[8] = {};
  float mrun[8], lrun[8];
#pragma unroll
  for (int v = 0; v < 8; ++v) { mrun[v] = -1e30f; lrun[v] = 0.0f; }

  u16_t* pbuf = plds[wv];
  const int ktLast = qLocal >> 5;

  for (int kt = 0; kt <= ktLast; ++kt) {
    const int kb = kt * 32;
    const u16_t* Kb0 = Kbf + ((size_t)b * SLEN + kb + n) * HDIM;
    __builtin_prefetch(Kb0 + 32 * HDIM, 0, 0);  // next key tile

    // Phase 1: issue all 8 K B-fragment loads (16 x b128)
    FragB kf[8];
#pragma unroll
    for (int c = 0; c < 4; ++c) {
      const u16_t* r0 = Kb0 + c * 32 + lh * 16;
      kf[c].u[0]     = *(const uint4*)(r0);
      kf[c].u[1]     = *(const uint4*)(r0 + 8);
      const u16_t* r1 = r0 + 16 * HDIM;
      kf[4 + c].u[0] = *(const uint4*)(r1);
      kf[4 + c].u[1] = *(const uint4*)(r1 + 8);
    }
    sched_split();

    // Phase 2: scores S[16q x 32k] = Q * K^T (two 16-col D tiles)
    v8f s0 = {}, s1 = {};
#pragma unroll
    for (int c = 0; c < 4; ++c) {
      s0 = bf_wmma(qf[c].v, kf[c].v, s0);
      s1 = bf_wmma(qf[c].v, kf[4 + c].v, s1);
    }
    sched_split();

    // Phase 3: issue V fragment loads NOW: independent of softmax, so the
    // exp/shuffle VALU work below hides their latency.
    FragB vf[8];
    const u16_t* Vb = Vt + (size_t)b * SLEN + kb + lh * 16;
#pragma unroll
    for (int nt = 0; nt < 8; ++nt) {
      const u16_t* vr = Vb + (size_t)(nt * 16 + n) * NROWS;
      vf[nt].u[0] = *(const uint4*)(vr);
      vf[nt].u[1] = *(const uint4*)(vr + 8);
    }
    sched_split();  // V loads stay above the softmax VALU block

    if (kt == ktLast) {  // causal mask, only diagonal tile
#pragma unroll
      for (int v = 0; v < 8; ++v) {
        const int q = qLocal + v + 8 * lh;
        if (kb + n > q)      s0[v] = -1e30f;
        if (kb + 16 + n > q) s1[v] = -1e30f;
      }
    }

    // Online softmax: row = VGPR index v + 8*lane_half, cols = 16 lanes
    float corr[8];
#pragma unroll
    for (int v = 0; v < 8; ++v) {
      float x = fmaxf(s0[v], s1[v]);
#pragma unroll
      for (int off = 8; off; off >>= 1) x = fmaxf(x, __shfl_xor(x, off, 32));
      const float mnew = fmaxf(mrun[v], x);
      const float c0 = __expf(mrun[v] - mnew);
      mrun[v] = mnew;
      const float p0 = __expf(s0[v] - mnew);
      const float p1 = __expf(s1[v] - mnew);
      s0[v] = p0; s1[v] = p1;
      float t = p0 + p1;
#pragma unroll
      for (int off = 8; off; off >>= 1) t += __shfl_xor(t, off, 32);
      lrun[v] = lrun[v] * c0 + t;
      corr[v] = c0;
    }
#pragma unroll
    for (int nt = 0; nt < 8; ++nt)
#pragma unroll
      for (int v = 0; v < 8; ++v) o[nt][v] *= corr[v];

    // Transpose P (q x k) through LDS into an A-fragment layout
#pragma unroll
    for (int v = 0; v < 8; ++v) {
      const int r = v + 8 * lh;
      pbuf[r * 40 + n]      = f2bf(s0[v]);
      pbuf[r * 40 + 16 + n] = f2bf(s1[v]);
    }
    asm volatile("s_wait_dscnt 0" ::: "memory");  // same-wave LDS RAW
    FragB pf;
    {
      const u16_t* pr = pbuf + n * 40 + lh * 16;
      pf.u[0] = *(const uint4*)(pr);        // k = lh*16 + 0..7
      pf.u[1] = *(const uint4*)(pr + 8);    // k = lh*16 + 8..15
    }

    // O += P(16x32) * V(32k x 128h); V fragments already in registers
#pragma unroll
    for (int nt = 0; nt < 8; ++nt)
      o[nt] = bf_wmma(pf.v, vf[nt].v, o[nt]);
  }

  // Epilogue: normalize and store f32
#pragma unroll
  for (int v = 0; v < 8; ++v) lrun[v] = 1.0f / lrun[v];
#pragma unroll
  for (int nt = 0; nt < 8; ++nt)
#pragma unroll
    for (int v = 0; v < 8; ++v)
      out[(rowG + v + 8 * lh) * HDIM + nt * 16 + n] = o[nt][v] * lrun[v];
}

extern "C" void kernel_launch(void* const* d_in, const int* in_sizes, int n_in,
                              void* d_out, int out_size, void* d_ws, size_t ws_size,
                              hipStream_t stream) {
  const float* E  = (const float*)d_in[0];   // [4,4096,1024]
  const float* Wk = (const float*)d_in[1];   // [128,1024]
  const float* Wq = (const float*)d_in[2];
  const float* Wv = (const float*)d_in[3];

  // Workspace layout (bytes), all 256-aligned; total ~46.9 MB
  char* ws = (char*)d_ws;
  u16_t* Ebf = (u16_t*)(ws + 0);                  // 16384*1024*2 = 33,554,432
  u16_t* Wbf = (u16_t*)(ws + 33554432);           // 3*128*1024*2 =    786,432
  u16_t* Qbf = (u16_t*)(ws + 33554432 + 786432);  // 16384*128*2  =  4,194,304
  u16_t* Kbf = (u16_t*)(ws + 33554432 + 786432 + 4194304);
  u16_t* Vt  = (u16_t*)(ws + 33554432 + 786432 + 2 * 4194304);
  float* out = (float*)d_out;                     // [4,4096,128] f32

  cvt_e_kernel<<<16384, 256, 0, stream>>>(E, Ebf);        // 16.7M elems, 4/thread
  cvt_w_kernel<<<512, 256, 0, stream>>>(Wk, Wq, Wv, Wbf); // 131072 elems/matrix
  proj_kernel<<<1024, 96, 0, stream>>>(Ebf, Wbf, Qbf, Kbf, Vt);
  attn_kernel<<<256, 128, 0, stream>>>(Qbf, Kbf, Vt, out);
}
